// GATDecoder_15221364097589
// MI455X (gfx1250) — matrix operand order repaired and checked by reading, not compiled, not run on hardware
//
#include <hip/hip_runtime.h>

// ---------------------------------------------------------------------------
// GATDecoder for MI455X (gfx1250): bf16 WMMA GEMMs + fused edge MLP,
// TDM/async-LDS staging, per-node flash cross-attention, atomic segment softmax.
// ---------------------------------------------------------------------------

typedef __attribute__((ext_vector_type(16))) __bf16 v16bf;
typedef __attribute__((ext_vector_type(8)))  __bf16 v8bf;
typedef __attribute__((ext_vector_type(4)))  __bf16 v4bf;
typedef __attribute__((ext_vector_type(8)))  float  v8f;

#if __has_builtin(__builtin_amdgcn_tensor_load_to_lds) && \
    __has_builtin(__builtin_amdgcn_s_wait_tensorcnt)
#define HAVE_TDM 1
#else
#define HAVE_TDM 0
#endif

#if __has_builtin(__builtin_amdgcn_global_load_async_to_lds_b128) && \
    __has_builtin(__builtin_amdgcn_s_wait_asynccnt)
#define HAVE_ASYNC 1
#else
#define HAVE_ASYNC 0
#endif

// Pointer types matching the async-LDS builtin signature:
// (v4i __global__*, v4i __shared__*, imm, imm)
typedef int v4i __attribute__((vector_size(16)));
typedef __attribute__((address_space(1))) v4i* gv4i_p;
typedef __attribute__((address_space(3))) v4i* lv4i_p;

typedef __attribute__((ext_vector_type(4))) unsigned tdm_g0_t;
typedef __attribute__((ext_vector_type(8))) int      tdm_g1_t;
typedef __attribute__((ext_vector_type(4))) int      tdm_g2_t;

union FragBF { v16bf v; v8bf h[2]; };

__device__ inline v16bf load_frag(const __bf16* lo, const __bf16* hi) {
  FragBF f;
  f.h[0] = *(const v8bf*)lo;
  f.h[1] = *(const v8bf*)hi;
  return f.v;
}

__device__ inline v8f zero8() {
  v8f c = {0.f, 0.f, 0.f, 0.f, 0.f, 0.f, 0.f, 0.f};
  return c;
}

// 32-byte global -> LDS stage (async path bypasses VGPRs, tracked by ASYNCcnt)
__device__ inline void stage32(const __bf16* g, __bf16* l) {
#if HAVE_ASYNC
  __builtin_amdgcn_global_load_async_to_lds_b128((gv4i_p)g, (lv4i_p)l, 0, 0);
  __builtin_amdgcn_global_load_async_to_lds_b128((gv4i_p)(g + 8), (lv4i_p)(l + 8),
                                                 0, 0);
#else
  *(v8bf*)l = *(const v8bf*)g;
  *(v8bf*)(l + 8) = *(const v8bf*)(g + 8);
#endif
}

__device__ inline void stage_fence() {
#if HAVE_ASYNC
  __builtin_amdgcn_s_wait_asynccnt(0);
#endif
  __syncthreads();
}

#if HAVE_TDM
// Issue a TDM 2D tile load: `rows` rows of `rowbytes` bytes (stride == rowbytes)
// from global `g` into LDS `l`.  Must be executed by exactly one wave.
__device__ inline void tdm_tile_load(const void* g, void* l, unsigned rowbytes,
                                     unsigned rows) {
  unsigned lds = (unsigned)(unsigned long long)(
      __attribute__((address_space(3))) char*)l;
  unsigned long long ga = (unsigned long long)g;
  tdm_g0_t g0;
  g0.x = 1u;                                    // count = 1 valid descriptor
  g0.y = lds;                                   // lds_addr
  g0.z = (unsigned)(ga & 0xffffffffull);        // global_addr[31:0]
  g0.w = (unsigned)((ga >> 32) & 0x01ffffffull) // global_addr[56:32]
         | (2u << 30);                          // type = 2 ("image")
  tdm_g1_t g1;
  g1[0] = 0;                                    // wgmask=0, data_size=1B
  g1[1] = (int)((rowbytes & 0xffffu) << 16);    // tensor_dim0[15:0]
  g1[2] = (int)((rowbytes >> 16) & 0xffffu);    // tensor_dim0[31:16], dim1 lo=0
  g1[3] = (int)(0x4000u | ((rowbytes & 0xffffu) << 16)); // dim1 hi (1<<30), tile_dim0
  g1[4] = (int)rows;                            // tile_dim1 = rows, tile_dim2 = 0
  g1[5] = (int)rowbytes;                        // tensor_dim0_stride[31:0]
  g1[6] = 0;
  g1[7] = 0;
  tdm_g2_t z4 = {0, 0, 0, 0};
  tdm_g1_t z8 = {0, 0, 0, 0, 0, 0, 0, 0};
  __builtin_amdgcn_tensor_load_to_lds(g0, g1, z4, z4, z8, 0);
}
#endif

// --------------------------- utility kernels -------------------------------

__global__ void k_b16(const float* __restrict__ s, __bf16* __restrict__ d, long n4) {
  long i = (long)blockIdx.x * 256 + threadIdx.x;
  if (i >= n4) return;
  float4 f = ((const float4*)s)[i];
  v4bf o;
  o.x = (__bf16)f.x;
  o.y = (__bf16)f.y;
  o.z = (__bf16)f.z;
  o.w = (__bf16)f.w;
  ((v4bf*)d)[i] = o;
}

__global__ void k_zero32(unsigned* __restrict__ p, long n) {
  long i = (long)blockIdx.x * 256 + threadIdx.x;
  if (i < n) p[i] = 0u;
}

// Pack W[K x 128] (row-major f32) into bf16 WMMA B-fragment order:
// fragment (kc, nt) -> 512 bf16 at ((kc*8+nt)*512), element = lane*16 + s,
// lane = 16*((kl>>3)&1) + col, s = (kl&7) + 8*(kl>>4) (table k order).
__global__ void k_pack(const float* __restrict__ W, __bf16* __restrict__ dst, int K) {
  int i = blockIdx.x * 256 + threadIdx.x;
  if (i >= K * 128) return;
  int k = i >> 7, n = i & 127;
  int kc = k >> 5, kl = k & 31, nt = n >> 4, nj = n & 15;
  dst[((kc * 8 + nt) << 9) + ((((kl >> 3) & 1) * 16 + nj) * 16) + (kl & 7) + ((kl >> 4) << 3)] = (__bf16)W[i];
}

// ----------------------------- GEMM (K=128) --------------------------------
// C[M x 128] = A_bf16[M x 128] @ Wpk.  Block = 256 thr = 8 waves, 32 rows
// (M must be a multiple of 32 -- true for all call sites).  A-tile staged by
// a single TDM descriptor when available.

template <bool BF16OUT>
__global__ __launch_bounds__(256) void k_gemm128_t(
    const __bf16* __restrict__ A, const __bf16* __restrict__ Wpk,
    void* __restrict__ C, int M) {
  __shared__ __bf16 As[32 * 128];
  const int m0 = blockIdx.x * 32;
  const int t = threadIdx.x;
#if HAVE_TDM
  if (t < 32) {
    tdm_tile_load(A + (long)m0 * 128, As, 256u, 32u);
    __builtin_amdgcn_s_wait_tensorcnt(0);
  }
  __syncthreads();
#else
  {
    int r = t >> 3, sg = t & 7;
    stage32(A + (long)(m0 + r) * 128 + sg * 16, &As[r * 128 + sg * 16]);
  }
  stage_fence();
#endif
  const int w = t >> 5, lane = t & 31;
  const int lm = lane & 15;
  const int koff = (lane < 16) ? 0 : 8;
  const int moff = (lane < 16) ? 0 : 8;
  const int col = w * 16 + lm;
  for (int rt = 0; rt < 2; ++rt) {
    v8f c = zero8();
#pragma unroll
    for (int kc = 0; kc < 4; ++kc) {
      const __bf16* ap = &As[(rt * 16 + lm) * 128 + kc * 32 + koff];
      v16bf a = load_frag(ap, ap + 16);
      const __bf16* bp = Wpk + ((kc * 8 + w) << 9) + lane * 16;
      v16bf b = load_frag(bp, bp + 8);
      c = __builtin_amdgcn_wmma_f32_16x16x32_bf16(false, a, false, b, (short)0, c,
                                                  false, false);
    }
    long row0 = m0 + rt * 16 + moff;
    if (BF16OUT) {
      __bf16* cp = (__bf16*)C + row0 * 128 + col;
#pragma unroll
      for (int i = 0; i < 8; ++i) cp[(long)i * 128] = (__bf16)c[i];
    } else {
      float* cp = (float*)C + row0 * 128 + col;
#pragma unroll
      for (int i = 0; i < 8; ++i) cp[(long)i * 128] = c[i];
    }
  }
}

// ------------------------ fused residual edge MLP --------------------------
// e += relu([e, x[src], x[dst]] @ Ew1 + Eb1) @ Ew2 + Eb2   (32 edges / block,
// E is a multiple of 32).  Gathered rows staged via async global->LDS.

__global__ __launch_bounds__(256) void k_edge_mlp(
    const __bf16* __restrict__ eb, const __bf16* __restrict__ xb,
    const int* __restrict__ src, const int* __restrict__ dst,
    const __bf16* __restrict__ w1pk, const float* __restrict__ b1,
    const __bf16* __restrict__ w2pk, const float* __restrict__ b2,
    float* __restrict__ e, int E) {
  __shared__ __bf16 Acat[32 * 384];  // 24 KB
  __shared__ __bf16 Hd[32 * 128];    //  8 KB
  const int e0 = blockIdx.x * 32;
  const int t = threadIdx.x;
  // stage [e | x[src] | x[dst]] (768 16-elem segments, 3 per thread)
#pragma unroll
  for (int si = 0; si < 3; ++si) {
    int lin = t + si * 256;
    int r = lin / 24, sg = lin % 24;
    int row = e0 + r;
    const __bf16* sp;
    if (sg < 8)       sp = eb + (long)row * 128 + sg * 16;
    else if (sg < 16) sp = xb + (long)src[row] * 128 + (sg - 8) * 16;
    else              sp = xb + (long)dst[row] * 128 + (sg - 16) * 16;
    stage32(sp, &Acat[r * 384 + sg * 16]);
  }
  stage_fence();
  const int w = t >> 5, lane = t & 31;
  const int lm = lane & 15;
  const int koff = (lane < 16) ? 0 : 8;
  const int moff = (lane < 16) ? 0 : 8;
  const int col = w * 16 + lm;
  // GEMM1: 32x384 @ 384x128 -> relu(+b1) -> Hd (bf16)
  for (int rt = 0; rt < 2; ++rt) {
    v8f c = zero8();
#pragma unroll
    for (int kc = 0; kc < 12; ++kc) {
      const __bf16* ap = &Acat[(rt * 16 + lm) * 384 + kc * 32 + koff];
      v16bf a = load_frag(ap, ap + 16);
      const __bf16* bp = w1pk + ((kc * 8 + w) << 9) + lane * 16;
      v16bf b = load_frag(bp, bp + 8);
      c = __builtin_amdgcn_wmma_f32_16x16x32_bf16(false, a, false, b, (short)0, c,
                                                  false, false);
    }
    float bias = b1[col];
    __bf16* hp = &Hd[(rt * 16 + moff) * 128 + col];
#pragma unroll
    for (int i = 0; i < 8; ++i) hp[i * 128] = (__bf16)fmaxf(c[i] + bias, 0.f);
  }
  __syncthreads();
  // GEMM2: 32x128 @ 128x128, residual into e (in place)
  for (int rt = 0; rt < 2; ++rt) {
    v8f c = zero8();
#pragma unroll
    for (int kc = 0; kc < 4; ++kc) {
      const __bf16* ap = &Hd[(rt * 16 + lm) * 128 + kc * 32 + koff];
      v16bf a = load_frag(ap, ap + 16);
      const __bf16* bp = w2pk + ((kc * 8 + w) << 9) + lane * 16;
      v16bf b = load_frag(bp, bp + 8);
      c = __builtin_amdgcn_wmma_f32_16x16x32_bf16(false, a, false, b, (short)0, c,
                                                  false, false);
    }
    float bias = b2[col];
    float* ep = e + (long)(e0 + rt * 16 + moff) * 128 + col;
#pragma unroll
    for (int i = 0; i < 8; ++i) ep[(long)i * 128] = ep[(long)i * 128] + c[i] + bias;
  }
}

// ------------------------- GAT attention pieces ----------------------------

__global__ void k_sdot_node(const __bf16* __restrict__ hb,
                            const float* __restrict__ as, const float* __restrict__ ad,
                            float* __restrict__ ssrc, float* __restrict__ sdst, int N) {
  int t = blockIdx.x * 256 + threadIdx.x;
  if (t >= N * 4) return;
  int n = t >> 2, h = t & 3;
  const __bf16* hp = hb + (long)n * 128 + h * 32;
  const float* a1 = as + h * 32;
  const float* a2 = ad + h * 32;
  float s1 = 0.f, s2 = 0.f;
#pragma unroll
  for (int c = 0; c < 32; ++c) {
    float hv = (float)hp[c];
    s1 += hv * a1[c];
    s2 += hv * a2[c];
  }
  ssrc[t] = s1;
  sdst[t] = s2;
}

__global__ void k_sdot_edge(const __bf16* __restrict__ ehb,
                            const float* __restrict__ ae,
                            float* __restrict__ sedge, int E) {
  int t = blockIdx.x * 256 + threadIdx.x;
  if (t >= E * 4) return;
  int e = t >> 2, h = t & 3;
  const __bf16* ep = ehb + (long)e * 128 + h * 32;
  const float* a = ae + h * 32;
  float s = 0.f;
#pragma unroll
  for (int c = 0; c < 32; ++c) s += (float)ep[c] * a[c];
  sedge[t] = s;
}

__device__ inline unsigned ford(float f) {
  unsigned b = __float_as_uint(f);
  return (b & 0x80000000u) ? ~b : (b | 0x80000000u);
}
__device__ inline float funord(unsigned u) {
  return __uint_as_float((u & 0x80000000u) ? (u ^ 0x80000000u) : ~u);
}

__global__ void k_logits(const int* __restrict__ src, const int* __restrict__ dst,
                         const float* __restrict__ ssrc, const float* __restrict__ sdst,
                         const float* __restrict__ sedge, float* __restrict__ lg,
                         unsigned* __restrict__ mx, int E) {
  int t = blockIdx.x * 256 + threadIdx.x;
  if (t >= E * 4) return;
  int e = t >> 2, h = t & 3;
  float v = ssrc[src[e] * 4 + h] + sdst[dst[e] * 4 + h] + sedge[t];
  v = (v > 0.f) ? v : 0.2f * v;  // leaky_relu
  lg[t] = v;
  atomicMax(&mx[dst[e] * 4 + h], ford(v));
}

__global__ void k_expw(const int* __restrict__ dst, const unsigned* __restrict__ mx,
                       float* __restrict__ lg, float* __restrict__ den, int E) {
  int t = blockIdx.x * 256 + threadIdx.x;
  if (t >= E * 4) return;
  int e = t >> 2, h = t & 3;
  int d = dst[e];
  float w = __expf(lg[t] - funord(mx[d * 4 + h]));
  lg[t] = w;
  atomicAdd(&den[d * 4 + h], w);
}

__global__ void k_conv(const int* __restrict__ src, const int* __restrict__ dst,
                       const float* __restrict__ wgt, const float* __restrict__ den,
                       const __bf16* __restrict__ hb, const __bf16* __restrict__ ehb,
                       float* __restrict__ conv, int E) {
  int t = blockIdx.x * 256 + threadIdx.x;
  if (t >= E * 4) return;
  int e = t >> 2, h = t & 3;
  int d = dst[e], s = src[e];
  float alpha = wgt[t] / (den[d * 4 + h] + 1e-16f);
  const __bf16* hp = hb + (long)s * 128 + h * 32;
  const __bf16* ep = ehb + (long)e * 128 + h * 32;
  float* cp = conv + (long)d * 128 + h * 32;
#pragma unroll
  for (int c = 0; c < 32; ++c)
    atomicAdd(cp + c, alpha * ((float)hp[c] + (float)ep[c]));
}

// ----------------------------- layer norms ---------------------------------
// One wave per node, 4 channels / lane.

__global__ __launch_bounds__(256) void k_ln1(
    const float* __restrict__ conv, const float* __restrict__ bn,
    const float* __restrict__ g, const float* __restrict__ bb,
    float* __restrict__ x, int N) {
  int node = (blockIdx.x * 256 + threadIdx.x) >> 5;
  if (node >= N) return;
  int lane = threadIdx.x & 31;
  float4 cv = *(const float4*)(conv + (long)node * 128 + lane * 4);
  float4 bn4 = *(const float4*)(bn + lane * 4);
  float v0 = cv.x + bn4.x, v1 = cv.y + bn4.y, v2 = cv.z + bn4.z, v3 = cv.w + bn4.w;
  float s = v0 + v1 + v2 + v3;
  for (int o = 16; o > 0; o >>= 1) s += __shfl_xor(s, o, 32);
  float mu = s * (1.f / 128.f);
  float d0 = v0 - mu, d1 = v1 - mu, d2 = v2 - mu, d3 = v3 - mu;
  float vs = d0 * d0 + d1 * d1 + d2 * d2 + d3 * d3;
  for (int o = 16; o > 0; o >>= 1) vs += __shfl_xor(vs, o, 32);
  float rs = rsqrtf(vs * (1.f / 128.f) + 1e-5f);
  float4 g4 = *(const float4*)(g + lane * 4);
  float4 b4 = *(const float4*)(bb + lane * 4);
  float4 xo = *(float4*)(x + (long)node * 128 + lane * 4);
  float4 out;
  out.x = fmaxf(d0 * rs * g4.x + b4.x, 0.f) + xo.x;
  out.y = fmaxf(d1 * rs * g4.y + b4.y, 0.f) + xo.y;
  out.z = fmaxf(d2 * rs * g4.z + b4.z, 0.f) + xo.z;
  out.w = fmaxf(d3 * rs * g4.w + b4.w, 0.f) + xo.w;
  *(float4*)(x + (long)node * 128 + lane * 4) = out;
}

__global__ __launch_bounds__(256) void k_ln2(
    const float* __restrict__ proj,
    const float* __restrict__ g, const float* __restrict__ bb,
    float* __restrict__ x, int N) {
  int node = (blockIdx.x * 256 + threadIdx.x) >> 5;
  if (node >= N) return;
  int lane = threadIdx.x & 31;
  float4 pv = *(const float4*)(proj + (long)node * 128 + lane * 4);
  float4 xo = *(float4*)(x + (long)node * 128 + lane * 4);
  float v0 = pv.x + xo.x, v1 = pv.y + xo.y, v2 = pv.z + xo.z, v3 = pv.w + xo.w;
  float s = v0 + v1 + v2 + v3;
  for (int o = 16; o > 0; o >>= 1) s += __shfl_xor(s, o, 32);
  float mu = s * (1.f / 128.f);
  float d0 = v0 - mu, d1 = v1 - mu, d2 = v2 - mu, d3 = v3 - mu;
  float vs = d0 * d0 + d1 * d1 + d2 * d2 + d3 * d3;
  for (int o = 16; o > 0; o >>= 1) vs += __shfl_xor(vs, o, 32);
  float rs = rsqrtf(vs * (1.f / 128.f) + 1e-5f);
  float4 g4 = *(const float4*)(g + lane * 4);
  float4 b4 = *(const float4*)(bb + lane * 4);
  float4 out;
  out.x = fmaxf(d0 * rs * g4.x + b4.x, 0.f);
  out.y = fmaxf(d1 * rs * g4.y + b4.y, 0.f);
  out.z = fmaxf(d2 * rs * g4.z + b4.z, 0.f);
  out.w = fmaxf(d3 * rs * g4.w + b4.w, 0.f);
  *(float4*)(x + (long)node * 128 + lane * 4) = out;
}

// ------------------- per-node flash cross-attention ------------------------
// One wave per valid node (padded dense rows never reach the output, so the
// B x MAX_NODE scatter is skipped entirely).  lane = channel, shfl-reduce per key.

__global__ __launch_bounds__(256) void k_attn(
    const __bf16* __restrict__ qb, const __bf16* __restrict__ kb,
    const __bf16* __restrict__ vb, const int* __restrict__ node2batch,
    __bf16* __restrict__ aob, int N) {
  int node = (blockIdx.x * 256 + threadIdx.x) >> 5;
  if (node >= N) return;
  int lane = threadIdx.x & 31;
  int b = node2batch[node];
  const __bf16* Kp = kb + (long)b * 256 * 128;
  const __bf16* Vp = vb + (long)b * 256 * 128;
  const float scale = 0.17677669529663687f;  // 1/sqrt(32)
  for (int h = 0; h < 4; ++h) {
    float qc = (float)qb[(long)node * 128 + h * 32 + lane];
    float m = -1e30f, lsum = 0.f, acc = 0.f;
    for (int key = 0; key < 256; ++key) {
      if ((key & 31) == 0 && key + 32 < 256) {
        __builtin_prefetch((const void*)(Kp + (long)(key + 32) * 128 + h * 32 + lane), 0, 1);
        __builtin_prefetch((const void*)(Vp + (long)(key + 32) * 128 + h * 32 + lane), 0, 1);
      }
      float p = qc * (float)Kp[(long)key * 128 + h * 32 + lane];
      for (int o = 16; o > 0; o >>= 1) p += __shfl_xor(p, o, 32);
      p *= scale;
      float mn = fmaxf(m, p);
      float corr = __expf(m - mn);
      float wgt = __expf(p - mn);
      float vv = (float)Vp[(long)key * 128 + h * 32 + lane];
      lsum = lsum * corr + wgt;
      acc = acc * corr + wgt * vv;
      m = mn;
    }
    aob[(long)node * 128 + h * 32 + lane] = (__bf16)(acc / (lsum + 1e-16f));
  }
}

// ------------------------------- host side ---------------------------------

extern "C" void kernel_launch(void* const* d_in, const int* in_sizes, int n_in,
                              void* d_out, int out_size, void* d_ws, size_t ws_size,
                              hipStream_t stream) {
  const int N = in_sizes[0] / 128;   // 24320
  const int E = in_sizes[1] / 128;   // 389120
  const int BM = in_sizes[2] / 128;  // 8192 = B * MEM_LEN
  const int L = 4;

  const float* node_feats = (const float*)d_in[0];
  const float* edge_feats = (const float*)d_in[1];
  const float* memory     = (const float*)d_in[2];
  const float* Wn  = (const float*)d_in[3];
  const float* bn  = (const float*)d_in[4];
  const float* We  = (const float*)d_in[5];
  const float* att_src  = (const float*)d_in[6];
  const float* att_dst  = (const float*)d_in[7];
  const float* att_edge = (const float*)d_in[8];
  const float* ln1g = (const float*)d_in[9];
  const float* ln1b = (const float*)d_in[10];
  const float* ln2g = (const float*)d_in[11];
  const float* ln2b = (const float*)d_in[12];
  const float* Wq = (const float*)d_in[13];
  const float* Wk = (const float*)d_in[14];
  const float* Wv = (const float*)d_in[15];
  const float* Wo = (const float*)d_in[16];
  const float* Ew1 = (const float*)d_in[17];
  const float* Eb1 = (const float*)d_in[18];
  const float* Ew2 = (const float*)d_in[19];
  const float* Eb2 = (const float*)d_in[20];
  const int* eidx = (const int*)d_in[21];
  const int* srcI = eidx;
  const int* dstI = eidx + E;
  const int* n2b  = (const int*)d_in[22];

  float* x = (float*)d_out;                 // N x 128
  float* e = (float*)d_out + (long)N * 128; // E x 128

  // ---- workspace carve (256B aligned) ----
  char* wp = (char*)d_ws;
  auto carve = [&](size_t bytes) -> void* {
    void* r = (void*)wp;
    wp += (bytes + 255) & ~(size_t)255;
    return r;
  };
  const long PKL = 163840;  // packed bf16 elems per layer
  __bf16* pk   = (__bf16*)carve((size_t)PKL * L * sizeof(__bf16));
  __bf16* xb   = (__bf16*)carve((size_t)N * 128 * 2);
  __bf16* ebuf = (__bf16*)carve((size_t)E * 128 * 2);
  __bf16* mb   = (__bf16*)carve((size_t)BM * 128 * 2);
  __bf16* hb   = (__bf16*)carve((size_t)N * 128 * 2);
  __bf16* ehb  = (__bf16*)carve((size_t)E * 128 * 2);
  __bf16* qb   = (__bf16*)carve((size_t)N * 128 * 2);
  __bf16* kb   = (__bf16*)carve((size_t)BM * 128 * 2);
  __bf16* vb   = (__bf16*)carve((size_t)BM * 128 * 2);
  __bf16* aob  = (__bf16*)carve((size_t)N * 128 * 2);
  float* ssrc  = (float*)carve((size_t)N * 4 * 4);
  float* sdst  = (float*)carve((size_t)N * 4 * 4);
  float* sedge = (float*)carve((size_t)E * 4 * 4);
  float* lg    = (float*)carve((size_t)E * 4 * 4);
  unsigned* mx = (unsigned*)carve((size_t)N * 4 * 4);
  float* den   = (float*)carve((size_t)N * 4 * 4);
  float* conv  = (float*)carve((size_t)N * 128 * 4);
  float* proj  = (float*)carve((size_t)N * 128 * 4);

  auto g1 = [](long n) { return dim3((unsigned)((n + 255) / 256)); };
  dim3 blk(256);

  // ---- init outputs (residual streams live in d_out) ----
  (void)hipMemcpyAsync(x, node_feats, (size_t)N * 128 * 4,
                       hipMemcpyDeviceToDevice, stream);
  (void)hipMemcpyAsync(e, edge_feats, (size_t)E * 128 * 4,
                       hipMemcpyDeviceToDevice, stream);

  // ---- pack all weights to bf16 WMMA fragment order ----
  for (int l = 0; l < L; ++l) {
    __bf16* pl = pk + (long)l * PKL;
    k_pack<<<g1(16384), blk, 0, stream>>>(Wn  + (long)l * 16384, pl + 0,      128);
    k_pack<<<g1(16384), blk, 0, stream>>>(We  + (long)l * 16384, pl + 16384,  128);
    k_pack<<<g1(16384), blk, 0, stream>>>(Wq  + (long)l * 16384, pl + 32768,  128);
    k_pack<<<g1(16384), blk, 0, stream>>>(Wk  + (long)l * 16384, pl + 49152,  128);
    k_pack<<<g1(16384), blk, 0, stream>>>(Wv  + (long)l * 16384, pl + 65536,  128);
    k_pack<<<g1(16384), blk, 0, stream>>>(Wo  + (long)l * 16384, pl + 81920,  128);
    k_pack<<<g1(49152), blk, 0, stream>>>(Ew1 + (long)l * 49152, pl + 98304,  384);
    k_pack<<<g1(16384), blk, 0, stream>>>(Ew2 + (long)l * 16384, pl + 147456, 128);
  }
  k_b16<<<g1((long)BM * 32), blk, 0, stream>>>(memory, mb, (long)BM * 32);

  for (int l = 0; l < L; ++l) {
    const __bf16* pl = pk + (long)l * PKL;
    // ---- GAT conv ----
    k_b16<<<g1((long)N * 32), blk, 0, stream>>>(x, xb, (long)N * 32);
    k_b16<<<g1((long)E * 32), blk, 0, stream>>>(e, ebuf, (long)E * 32);
    k_gemm128_t<true><<<dim3((N + 31) / 32), blk, 0, stream>>>(xb, pl + 0, hb, N);
    k_gemm128_t<true><<<dim3((E + 31) / 32), blk, 0, stream>>>(ebuf, pl + 16384, ehb, E);
    k_sdot_node<<<g1((long)N * 4), blk, 0, stream>>>(
        hb, att_src + (long)l * 128, att_dst + (long)l * 128, ssrc, sdst, N);
    k_sdot_edge<<<g1((long)E * 4), blk, 0, stream>>>(
        ehb, att_edge + (long)l * 128, sedge, E);
    k_zero32<<<g1((long)N * 4), blk, 0, stream>>>(mx, (long)N * 4);
    k_zero32<<<g1((long)N * 4), blk, 0, stream>>>((unsigned*)den, (long)N * 4);
    k_zero32<<<g1((long)N * 128), blk, 0, stream>>>((unsigned*)conv, (long)N * 128);
    k_logits<<<g1((long)E * 4), blk, 0, stream>>>(srcI, dstI, ssrc, sdst, sedge, lg, mx, E);
    k_expw<<<g1((long)E * 4), blk, 0, stream>>>(dstI, mx, lg, den, E);
    k_conv<<<g1((long)E * 4), blk, 0, stream>>>(srcI, dstI, lg, den, hb, ehb, conv, E);
    k_ln1<<<g1((long)N * 32), blk, 0, stream>>>(
        conv, bn + (long)l * 128, ln1g + (long)l * 128, ln1b + (long)l * 128, x, N);
    // ---- cross attention (per node; dense scatter eliminated) ----
    k_b16<<<g1((long)N * 32), blk, 0, stream>>>(x, xb, (long)N * 32);
    k_gemm128_t<true><<<dim3((N + 31) / 32), blk, 0, stream>>>(xb, pl + 32768, qb, N);
    k_gemm128_t<true><<<dim3((BM + 31) / 32), blk, 0, stream>>>(mb, pl + 49152, kb, BM);
    k_gemm128_t<true><<<dim3((BM + 31) / 32), blk, 0, stream>>>(mb, pl + 65536, vb, BM);
    k_attn<<<g1((long)N * 32), blk, 0, stream>>>(qb, kb, vb, n2b, aob, N);
    k_gemm128_t<false><<<dim3((N + 31) / 32), blk, 0, stream>>>(aob, pl + 81920, proj, N);
    k_ln2<<<g1((long)N * 32), blk, 0, stream>>>(
        proj, ln2g + (long)l * 128, ln2b + (long)l * 128, x, N);
    // ---- fused residual edge MLP (uses pre-update ebuf, post-update xb) ----
    k_b16<<<g1((long)N * 32), blk, 0, stream>>>(x, xb, (long)N * 32);
    k_edge_mlp<<<dim3((E + 31) / 32), blk, 0, stream>>>(
        ebuf, xb, srcI, dstI, pl + 98304, Eb1 + (long)l * 128,
        pl + 147456, Eb2 + (long)l * 128, e, E);
  }
}
